// AdaptivePool_38697655337319
// MI455X (gfx1250) — matrix operand order repaired and checked by reading, not compiled, not run on hardware
//
#include <hip/hip_runtime.h>
#include <math.h>

// CDNA5 / gfx1250 fused kernel: cross-modal attention pooling + gated
// per-center cosine retrieval. Wave32, v_wmma_f32_16x16x32_f16.

typedef __attribute__((ext_vector_type(16))) _Float16 v16h;
typedef __attribute__((ext_vector_type(8)))  float    v8f;

union FragU {
    v16h h;
    uint4 q[2];          // 16 halfs = 2 x 16B
};

union H4 {
    _Float16 h[4];
    uint2 u;             // 4 halfs = 8B
};

#define A_TOT 256
#define B_TOT 256
#define V_FR  12
#define D_DIM 512
#define C_CTR 8
#define W_CH  64     // D/C
#define H_HID 256    // 4W
#define K_IN  128    // 2W
#define TEMP_ 5.0f

#define TA 16
#define TB 16
#define NBV (TB * V_FR)   // 192

#define F_SS    (TA * NBV)        // 3072 f32 : S then softmax weights
#define F_STN   (TA * C_CTR)      // 128
#define F_GATE  256
#define F_NUM   256
#define F_SQ    256
#define F_OUT   256
#define F_W2    256
#define F_B1    256
#define N_FLOATS (F_SS + F_STN + F_GATE + F_NUM + F_SQ + F_OUT + F_W2 + F_B1)

#define H_TEXT  (TA * D_DIM)      // 8192  halfs
#define H_W1T   (H_HID * K_IN)    // 32768 halfs (W1^T: [n][k])
#define H_VF    (256 * W_CH)      // 16384 halfs
#define H_VT    (NBV * W_CH)      // 12288 halfs
#define N_HALFS (H_TEXT + H_W1T + H_VF + H_VT)

#define SMEM_BYTES (N_FLOATS * 4 + N_HALFS * 2)  // 158208 B

__global__ __launch_bounds__(256)
void fused_retrieval_kernel(const float* __restrict__ text,
                            const float* __restrict__ video,
                            const float* __restrict__ W1,
                            const float* __restrict__ b1,
                            const float* __restrict__ W2,
                            const float* __restrict__ b2,
                            float* __restrict__ out)
{
    extern __shared__ unsigned char smem_raw[];
    float* sS    = (float*)smem_raw;
    float* sTn   = sS    + F_SS;
    float* sGate = sTn   + F_STN;
    float* sNum  = sGate + F_GATE;
    float* sSq   = sNum  + F_NUM;
    float* sOut  = sSq   + F_SQ;
    float* sW2   = sOut  + F_OUT;
    float* sB1   = sW2   + F_W2;
    _Float16* sTextH = (_Float16*)(sB1 + F_B1);
    _Float16* sW1T   = sTextH + H_TEXT;
    _Float16* sVf    = sW1T   + H_W1T;
    _Float16* sVt    = sVf    + H_VF;

    const int tid  = threadIdx.x;
    const int lane = tid & 31;
    const int wave = tid >> 5;
    const int a0 = blockIdx.x * TA;
    const int b0 = blockIdx.y * TB;
    const int hi = lane >> 4;
    const int ln = lane & 15;
    const int kbh = hi ? 4 : 0;        // A-frag dword base (kb/2)

    // ---- setup: W1^T, W2, b1, text f16, text-chunk inverse norms ----
    #pragma unroll 4
    for (int i = 0; i < 32; ++i) {             // 8192 quads of W1
        int q = i * 256 + tid;
        int k = q >> 6, n4 = (q & 63) * 4;
        float4 w4 = *(const float4*)(W1 + k * H_HID + n4);
        sW1T[(n4 + 0) * K_IN + k] = (_Float16)w4.x;
        sW1T[(n4 + 1) * K_IN + k] = (_Float16)w4.y;
        sW1T[(n4 + 2) * K_IN + k] = (_Float16)w4.z;
        sW1T[(n4 + 3) * K_IN + k] = (_Float16)w4.w;
    }
    sW2[tid] = W2[tid];
    sB1[tid] = b1[tid];
    #pragma unroll 4
    for (int i = 0; i < 8; ++i) {              // 2048 quads of text
        int q = i * 256 + tid;
        int a = q >> 7, d4 = (q & 127) * 4;
        float4 t4 = *(const float4*)(text + (a0 + a) * D_DIM + d4);
        H4 h4;
        h4.h[0] = (_Float16)t4.x; h4.h[1] = (_Float16)t4.y;
        h4.h[2] = (_Float16)t4.z; h4.h[3] = (_Float16)t4.w;
        *(uint2*)(sTextH + a * D_DIM + d4) = h4.u;
    }
    if (tid < TA * C_CTR) {
        int a = tid >> 3, c = tid & 7;
        float s = 0.f;
        for (int d = 0; d < W_CH; ++d) {
            float tv = text[(a0 + a) * D_DIM + c * W_CH + d];
            s += tv * tv;
        }
        sTn[tid] = rsqrtf(s);
    }
    sOut[tid] = 0.f;
    const float b2v = b2[0];

    // ---- GEMM1: S[16a x 192bv] = text16 @ video^T, K=512 ----
    v8f acc0 = {};
    v8f acc1 = {};
    const bool two = (wave < 4);
    const int nt0 = wave, nt1 = wave + 8;

    for (int ks = 0; ks < 16; ++ks) {
        const int k0 = ks * 32;
        __syncthreads();
        #pragma unroll 3
        for (int i = 0; i < 6; ++i) {          // 1536 quads, 32-wide rows
            int q = i * 256 + tid;
            int r = q >> 3, c4 = (q & 7) * 4;
            int b = r / 12, v = r - b * 12;
            const float* gp = video + (((b0 + b) * V_FR) + v) * D_DIM + k0 + c4;
            float4 v4 = *(const float4*)gp;
            if (ks < 15) __builtin_prefetch(gp + 32, 0, 1);
            H4 h4;
            h4.h[0] = (_Float16)v4.x; h4.h[1] = (_Float16)v4.y;
            h4.h[2] = (_Float16)v4.z; h4.h[3] = (_Float16)v4.w;
            *(uint2*)(sVt + r * 32 + c4) = h4.u;
        }
        __syncthreads();

        FragU af;
        {   // A-frag: 16x32 f16 layout (M = ln, K interleave per ISA)
            const unsigned int* arow =
                (const unsigned int*)(sTextH + ln * D_DIM + k0);
            af.q[0] = *(const uint4*)(arow + kbh);
            af.q[1] = *(const uint4*)(arow + 8 + kbh);
        }
        {
            FragU bf;
            const unsigned int* brow =
                (const unsigned int*)(sVt + (nt0 * 16 + ln) * 32 + hi * 8);
            bf.q[0] = *(const uint4*)brow;
            bf.q[1] = *(const uint4*)(brow + 8);
            acc0 = __builtin_amdgcn_wmma_f32_16x16x32_f16(
                false, af.h, false, bf.h, (short)0, acc0, false, false);
        }
        if (two) {
            FragU bf;
            const unsigned int* brow =
                (const unsigned int*)(sVt + (nt1 * 16 + ln) * 32 + hi * 8);
            bf.q[0] = *(const uint4*)brow;
            bf.q[1] = *(const uint4*)(brow + 8);
            acc1 = __builtin_amdgcn_wmma_f32_16x16x32_f16(
                false, af.h, false, bf.h, (short)0, acc1, false, false);
        }
    }
    #pragma unroll
    for (int r = 0; r < 8; ++r)
        sS[(r + hi * 8) * NBV + nt0 * 16 + ln] = acc0[r];
    if (two) {
        #pragma unroll
        for (int r = 0; r < 8; ++r)
            sS[(r + hi * 8) * NBV + nt1 * 16 + ln] = acc1[r];
    }
    __syncthreads();

    // ---- softmax over V=12 (temp 5), in place ----
    {
        int a = tid >> 4, b = tid & 15;
        float* row = sS + a * NBV + b * V_FR;
        float m = row[0];
        #pragma unroll
        for (int v = 1; v < V_FR; ++v) m = fmaxf(m, row[v]);
        float e[V_FR], sum = 0.f;
        #pragma unroll
        for (int v = 0; v < V_FR; ++v) { e[v] = expf((row[v] - m) * (1.0f / TEMP_)); sum += e[v]; }
        float inv = 1.0f / sum;
        #pragma unroll
        for (int v = 0; v < V_FR; ++v) row[v] = e[v] * inv;
    }

    // ---- preload MLP B-fragments (W1^T) into registers: invariant over c ----
    // wave owns N-tiles nt0 = wave, nt1 = wave + 8.
    FragU bfr[8];                       // [t2*4 + ks]
    #pragma unroll
    for (int t2 = 0; t2 < 2; ++t2) {
        const int nt = wave + t2 * 8;
        #pragma unroll
        for (int ks = 0; ks < 4; ++ks) {
            const unsigned int* brow = (const unsigned int*)
                (sW1T + (nt * 16 + ln) * K_IN + ks * 32 + hi * 8);
            bfr[t2 * 4 + ks].q[0] = *(const uint4*)brow;
            bfr[t2 * 4 + ks].q[1] = *(const uint4*)(brow + 8);
        }
    }
    const float w2v0 = sW2[nt0 * 16 + ln];
    const float b1v0 = sB1[nt0 * 16 + ln];
    const float w2v1 = sW2[nt1 * 16 + ln];
    const float b1v1 = sB1[nt1 * 16 + ln];

    // ---- per-center loop ----
    for (int cc = 0; cc < C_CTR; ++cc) {
        const int cbase = cc * W_CH;
        __syncthreads();
        #pragma unroll 4
        for (int i = 0; i < 12; ++i) {         // 3072 quads, 64-wide rows
            int q = i * 256 + tid;
            int r = q >> 4, c4 = (q & 15) * 4;
            int b = r / 12, v = r - b * 12;
            const float* gp = video + (((b0 + b) * V_FR) + v) * D_DIM + cbase + c4;
            float4 v4 = *(const float4*)gp;
            if (cc < C_CTR - 1) __builtin_prefetch(gp + W_CH, 0, 1);
            H4 h4;
            h4.h[0] = (_Float16)v4.x; h4.h[1] = (_Float16)v4.y;
            h4.h[2] = (_Float16)v4.z; h4.h[3] = (_Float16)v4.w;
            *(uint2*)(sVt + r * 64 + c4) = h4.u;
        }
        __syncthreads();

        // phase B: vf = sum_v w * video-chunk; cosine partials; gate init
        {
            int p = tid, a = p >> 4, b = p & 15;
            float w12[V_FR];
            #pragma unroll
            for (int v = 0; v < V_FR; ++v) w12[v] = sS[a * NBV + b * V_FR + v];
            float numer = 0.f, sq = 0.f;
            const _Float16* vrows = sVt + b * V_FR * 64;
            for (int d = 0; d < W_CH; ++d) {
                float vf = 0.f;
                #pragma unroll
                for (int v = 0; v < V_FR; ++v)
                    vf += w12[v] * (float)vrows[v * 64 + d];
                sVf[p * W_CH + d] = (_Float16)vf;
                float tf = (float)sTextH[a * D_DIM + cbase + d];
                numer += tf * vf;
                sq += vf * vf;
            }
            sNum[p] = numer;
            sSq[p]  = sq;
            sGate[p] = b2v;
        }
        __syncthreads();

        // phase C: gate += relu([t|vf] @ W1 + b1) @ W2 via WMMA,
        // B-frags live in registers; loop over all 16 M-tiles.
        for (int mt = 0; mt < 16; ++mt) {
            const int m0 = mt * 16;
            v8f accA = {};
            v8f accB = {};
            #pragma unroll
            for (int ks = 0; ks < 4; ++ks) {
                const int k0 = ks * 32;
                FragU af;
                const int pm = m0 + ln;
                const unsigned int* arow = (k0 < 64)
                    ? (const unsigned int*)(sTextH + (pm >> 4) * D_DIM + cbase + k0)
                    : (const unsigned int*)(sVf + pm * W_CH + (k0 - 64));
                af.q[0] = *(const uint4*)(arow + kbh);
                af.q[1] = *(const uint4*)(arow + 8 + kbh);
                accA = __builtin_amdgcn_wmma_f32_16x16x32_f16(
                    false, af.h, false, bfr[ks].h, (short)0, accA, false, false);
                accB = __builtin_amdgcn_wmma_f32_16x16x32_f16(
                    false, af.h, false, bfr[4 + ks].h, (short)0, accB, false, false);
            }
            // fused relu + b1 + W2-dot; 16-lane xor reduce; atomic into gate
            #pragma unroll
            for (int r = 0; r < 8; ++r) {
                float hA = accA[r] + b1v0;
                float hB = accB[r] + b1v1;
                hA = hA > 0.f ? hA : 0.f;
                hB = hB > 0.f ? hB : 0.f;
                float contrib = hA * w2v0 + hB * w2v1;
                contrib += __shfl_xor(contrib, 1, 32);
                contrib += __shfl_xor(contrib, 2, 32);
                contrib += __shfl_xor(contrib, 4, 32);
                contrib += __shfl_xor(contrib, 8, 32);
                if (ln == 0)
                    atomicAdd(&sGate[m0 + r + hi * 8], contrib);
            }
        }
        __syncthreads();

        // epilogue: out += gate * cos(t_c, vf_c)
        {
            int p = tid, a = p >> 4;
            float cosv = sNum[p] * rsqrtf(sSq[p]) * sTn[a * C_CTR + cc];
            sOut[p] += sGate[p] * cosv;
        }
    }

    out[(a0 + (tid >> 4)) * B_TOT + b0 + (tid & 15)] = sOut[tid];
}

extern "C" void kernel_launch(void* const* d_in, const int* in_sizes, int n_in,
                              void* d_out, int out_size, void* d_ws, size_t ws_size,
                              hipStream_t stream) {
    const float* text  = (const float*)d_in[0];
    const float* video = (const float*)d_in[1];
    const float* W1    = (const float*)d_in[2];
    const float* b1    = (const float*)d_in[3];
    const float* W2    = (const float*)d_in[4];
    const float* b2    = (const float*)d_in[5];
    float* out = (float*)d_out;

    dim3 grid(A_TOT / TA, B_TOT / TB);
    fused_retrieval_kernel<<<grid, 256, SMEM_BYTES, stream>>>(
        text, video, W1, b1, W2, b2, out);
}
